// Mamba2_29463475651260
// MI455X (gfx1250) — matrix operand order repaired and checked by reading, not compiled, not run on hardware
//
#include <hip/hip_runtime.h>
#include <hip/hip_bf16.h>

// ---------------- problem constants ----------------
#define D_MODEL   1024
#define D_INNER   2048
#define HEADDIM   64
#define NHEADS    32
#define D_STATE   128
#define D_CONV    4
#define CHUNK     256
#define BATCH     2
#define SEQLEN    4096
#define NCHUNK    (SEQLEN / CHUNK)                  // 16
#define D_IN_PROJ (2*D_INNER + 2*D_STATE + NHEADS)  // 4384
#define CONV_DIM  (D_INNER + 2*D_STATE)             // 2304
#define ROWS      (BATCH * SEQLEN)                  // 8192

// ---------------- WMMA tile config (wave32, 8 waves = 4x2) ----------------
#define BM 128
#define BN 64
#define BK 32
#define LDSA 40     // A tile row stride (80B: 16B-aligned, conflict-free)
#define LDSBT 40    // transposed B tile row stride
#define NTHREADS 256

typedef __bf16 bf16;
typedef __attribute__((ext_vector_type(16))) __bf16 v16bf;
typedef __attribute__((ext_vector_type(8)))  __bf16 v8bf;
typedef __attribute__((ext_vector_type(4)))  __bf16 v4bf;
typedef __attribute__((ext_vector_type(8)))  float  v8f;
typedef __attribute__((ext_vector_type(4)))  float  v4f;

union V16 { v16bf v; v8bf h[2]; };

__device__ __forceinline__ v4bf cvt4(v4f g) {
  v4bf o;
  o[0] = (bf16)g[0]; o[1] = (bf16)g[1]; o[2] = (bf16)g[2]; o[3] = (bf16)g[3];
  return o;
}

// ---- fragment loaders per CDNA5 ISA 7.12.2 VGPR layouts (wave32) ----
__device__ __forceinline__ v16bf ld_frag_A(const bf16* As, int m0) {
  const int lane = threadIdx.x & 31;
  const bf16* p = As + (m0 + (lane & 15)) * LDSA + 8 * (lane >> 4);
  V16 u;
  u.h[0] = *(const v8bf*)(p);        // k = 8h .. 8h+7
  u.h[1] = *(const v8bf*)(p + 16);   // k = 16+8h .. 16+8h+7
  return u.v;
}

__device__ __forceinline__ v16bf ld_frag_B(const bf16* Bst, int n0) {
  const int lane = threadIdx.x & 31;
  const bf16* p = Bst + (n0 + (lane & 15)) * LDSBT + 8 * (lane >> 4);
  V16 u;
  u.h[0] = *(const v8bf*)(p);
  u.h[1] = *(const v8bf*)(p + 16);
  return u.v;
}

__device__ __forceinline__ v8f wmma_bf16(v16bf a, v16bf b, v8f c) {
  return __builtin_amdgcn_wmma_f32_16x16x32_bf16(false, a, false, b, (short)0, c,
                                                 false, false);
}

__device__ __forceinline__ void mma_step(const bf16* As, const bf16* Bst, v8f acc[2][2]) {
  const int wid = threadIdx.x >> 5;
  const int wm = (wid >> 1) * 32;
  const int wn = (wid & 1) * 32;
  v16bf a0 = ld_frag_A(As, wm);
  v16bf a1 = ld_frag_A(As, wm + 16);
  v16bf b0 = ld_frag_B(Bst, wn);
  v16bf b1 = ld_frag_B(Bst, wn + 16);
  acc[0][0] = wmma_bf16(a0, b0, acc[0][0]);
  acc[0][1] = wmma_bf16(a0, b1, acc[0][1]);
  acc[1][0] = wmma_bf16(a1, b0, acc[1][0]);
  acc[1][1] = wmma_bf16(a1, b1, acc[1][1]);
}

__device__ __forceinline__ void store_acc(float* C, int ldc, int M, int N,
                                          int m_base, int n_base, v8f acc[2][2],
                                          int accumulate) {
  const int wid  = threadIdx.x >> 5;
  const int lane = threadIdx.x & 31;
  const int wm = (wid >> 1) * 32;
  const int wn = (wid & 1) * 32;
  const int nn = lane & 15;
  const int hh = lane >> 4;
#pragma unroll
  for (int i = 0; i < 2; ++i) {
#pragma unroll
    for (int jt = 0; jt < 2; ++jt) {
      const int n = n_base + wn + jt*16 + nn;
#pragma unroll
      for (int r = 0; r < 8; ++r) {
        const int m = m_base + wm + i*16 + r + 8*hh;
        if (m < M && n < N) {
          const float v = acc[i][jt][r];
          if (accumulate) C[(long long)m*ldc + n] += v;
          else            C[(long long)m*ldc + n]  = v;
        }
      }
    }
  }
}

// ---------------- generic fp32-in / bf16-WMMA / fp32-out GEMM ----------------
// Double-buffered LDS, software-pipelined global loads (1 barrier / k-step).
__global__ void k_gemm(const float* __restrict__ A, const float* __restrict__ B,
                       float* __restrict__ C,
                       int M, int N, int K, int lda, int ldb, int ldc,
                       long long sA, long long sB, long long sC,
                       int transB, int accumulate) {
  __shared__ alignas(16) bf16 As[2][BM][LDSA];
  __shared__ alignas(16) bf16 Bst[2][BN][LDSBT];
  const int nb = blockIdx.x * BN;
  const int mb = blockIdx.y * BM;
  A += (long long)blockIdx.z * sA;
  B += (long long)blockIdx.z * sB;
  C += (long long)blockIdx.z * sC;
  const bool fast = (mb + BM <= M) && (nb + BN <= N) && ((K & (BK-1)) == 0);
  v8f acc[2][2] = {};
  v4f ra[4], rb[2];

  auto load_regs = [&](int k0) {
#pragma unroll
    for (int i = 0; i < 4; ++i) {                    // A: BM x BK = 1024 v4f/4
      const int t = threadIdx.x + i*NTHREADS;
      const int r = t >> 3, c4 = (t & 7) * 4;
      ra[i] = *(const v4f*)&A[(long long)(mb + r)*lda + k0 + c4];
    }
    if (transB) {
#pragma unroll
      for (int i = 0; i < 2; ++i) {                  // B: [N][K]
        const int t = threadIdx.x + i*NTHREADS;
        const int n = t >> 3, k4 = (t & 7) * 4;
        rb[i] = *(const v4f*)&B[(long long)(nb + n)*ldb + k0 + k4];
      }
    } else {
#pragma unroll
      for (int i = 0; i < 2; ++i) {                  // B: [K][N]
        const int t = threadIdx.x + i*NTHREADS;
        const int r = t >> 4, n4 = (t & 15) * 4;
        rb[i] = *(const v4f*)&B[(long long)(k0 + r)*ldb + nb + n4];
      }
    }
  };

  auto store_regs = [&](int buf) {
#pragma unroll
    for (int i = 0; i < 4; ++i) {
      const int t = threadIdx.x + i*NTHREADS;
      const int r = t >> 3, c4 = (t & 7) * 4;
      *(v4bf*)&As[buf][r][c4] = cvt4(ra[i]);
    }
    if (transB) {
#pragma unroll
      for (int i = 0; i < 2; ++i) {
        const int t = threadIdx.x + i*NTHREADS;
        const int n = t >> 3, k4 = (t & 7) * 4;
        *(v4bf*)&Bst[buf][n][k4] = cvt4(rb[i]);
      }
    } else {
#pragma unroll
      for (int i = 0; i < 2; ++i) {
        const int t = threadIdx.x + i*NTHREADS;
        const int r = t >> 4, n4 = (t & 15) * 4;
        Bst[buf][n4+0][r] = (bf16)rb[i][0];
        Bst[buf][n4+1][r] = (bf16)rb[i][1];
        Bst[buf][n4+2][r] = (bf16)rb[i][2];
        Bst[buf][n4+3][r] = (bf16)rb[i][3];
      }
    }
  };

  if (fast) {
    load_regs(0);
    int buf = 0;
    for (int k0 = 0; k0 < K; k0 += BK, buf ^= 1) {
      store_regs(buf);
      __syncthreads();
      if (k0 + BK < K) {
        // prefetch two tiles ahead, then pipeline next tile's loads over MMA
        if (k0 + 2*BK < K)
          __builtin_prefetch(&A[(long long)(mb + (threadIdx.x >> 1))*lda
                                + k0 + 2*BK], 0, 1);
        load_regs(k0 + BK);
      }
      mma_step(&As[buf][0][0], &Bst[buf][0][0], acc);
    }
  } else {
    // scalar guarded path (ragged edge tiles only)
    for (int k0 = 0; k0 < K; k0 += BK) {
      for (int t = threadIdx.x; t < BM*BK; t += NTHREADS) {
        const int r = t >> 5, cc = t & 31;
        const int gm = mb + r, gk = k0 + cc;
        float v = 0.f;
        if (gm < M && gk < K) v = A[(long long)gm*lda + gk];
        As[0][r][cc] = (bf16)v;
      }
      for (int t = threadIdx.x; t < BK*BN; t += NTHREADS) {
        const int r = t >> 6, cc = t & 63;
        const int gk = k0 + r, gn = nb + cc;
        float v = 0.f;
        if (gk < K && gn < N)
          v = transB ? B[(long long)gn*ldb + gk] : B[(long long)gk*ldb + gn];
        Bst[0][cc][r] = (bf16)v;
      }
      __syncthreads();
      mma_step(&As[0][0][0], &Bst[0][0][0], acc);
      __syncthreads();
    }
  }
  store_acc(C, ldc, M, N, mb, nb, acc, accumulate);
}

// ---------------- conv1d + SiLU + softplus(dt) + dA + dt-scaled X ----------------
__global__ void k_convact(const float* __restrict__ zxbcdt,
                          const float* __restrict__ conv_w,
                          const float* __restrict__ conv_b,
                          const float* __restrict__ dt_bias,
                          const float* __restrict__ A_log,
                          float* __restrict__ xBC,
                          float* __restrict__ Xs,
                          float* __restrict__ dA) {
  const int row = blockIdx.x;            // b*SEQLEN + l
  const int l = row & (SEQLEN - 1);
  __shared__ float sdt[NHEADS];
  if (threadIdx.x < NHEADS) {
    const int h = threadIdx.x;
    const float x = zxbcdt[(long long)row*D_IN_PROJ + 2*D_INNER + 2*D_STATE + h]
                    + dt_bias[h];
    const float dt = (x > 20.f) ? x : log1pf(expf(x));      // softplus
    sdt[h] = dt;
    dA[row*NHEADS + h] = -expf(A_log[h]) * dt;
  }
  __syncthreads();
  for (int ch = threadIdx.x; ch < CONV_DIM; ch += NTHREADS) {
    float acc = conv_b[ch];
#pragma unroll
    for (int j = 0; j < D_CONV; ++j) {
      const int ls = l - (D_CONV - 1) + j;
      if (ls >= 0)
        acc += zxbcdt[((long long)(row - (D_CONV - 1) + j))*D_IN_PROJ + D_INNER + ch]
               * conv_w[ch*D_CONV + j];
    }
    const float v = acc / (1.f + expf(-acc));               // SiLU
    xBC[(long long)row*CONV_DIM + ch] = v;
    if (ch < D_INNER) Xs[(long long)row*D_INNER + ch] = v * sdt[ch >> 6];
  }
}

// ---------------- per-(b,chunk,head) cumulative sum of dA ----------------
__global__ void k_cumsum(const float* __restrict__ dA, float* __restrict__ Acs) {
  const int z = blockIdx.x;              // b*NCHUNK + c
  const int h = threadIdx.x;
  if (h < NHEADS) {
    float run = 0.f;
    const long long rowbase = (long long)z * CHUNK;
    for (int l = 0; l < CHUNK; ++l) {
      run += dA[(rowbase + l)*NHEADS + h];
      Acs[((long long)z*NHEADS + h)*CHUNK + l] = run;
    }
  }
}

// ---------------- Y_diag = (G .* exp(Acs[l]-Acs[s]) .* tril) @ X ----------------
__global__ void k_ydiag(const float* __restrict__ G, const float* __restrict__ Xs,
                        const float* __restrict__ Acs, float* __restrict__ Y) {
  __shared__ alignas(16) bf16 As[BM][LDSA];
  __shared__ alignas(16) bf16 Bst[BN][LDSBT];
  __shared__ float sAcs[CHUNK];
  const int mtile = blockIdx.x;          // 0..1
  const int zidx  = blockIdx.y;          // (b*NCHUNK + c)*NHEADS + h
  const int bc = zidx / NHEADS;
  const int h  = zidx % NHEADS;
  const int mb = mtile * BM;
  for (int t = threadIdx.x; t < CHUNK; t += NTHREADS)
    sAcs[t] = Acs[(long long)zidx*CHUNK + t];
  __syncthreads();
  const float* Gc = G + (long long)bc * CHUNK * CHUNK;
  const long long rowbase = (long long)bc * CHUNK;
  v8f acc[2][2] = {};
  for (int k0 = 0; k0 < CHUNK; k0 += BK) {
    v4f ga[4], gb[2];
#pragma unroll
    for (int i = 0; i < 4; ++i) {        // batch-issue global loads
      const int t = threadIdx.x + i*NTHREADS;
      const int r = t >> 3, c4 = (t & 7) * 4;
      ga[i] = *(const v4f*)&Gc[(mb + r)*CHUNK + k0 + c4];
    }
#pragma unroll
    for (int i = 0; i < 2; ++i) {
      const int t = threadIdx.x + i*NTHREADS;
      const int r = t >> 4, p4 = (t & 15) * 4;
      gb[i] = *(const v4f*)&Xs[(rowbase + k0 + r)*D_INNER + h*HEADDIM + p4];
    }
#pragma unroll
    for (int i = 0; i < 4; ++i) {
      const int t = threadIdx.x + i*NTHREADS;
      const int r = t >> 3, c4 = (t & 7) * 4;
      const int lg = mb + r, sg0 = k0 + c4;
      const float al = sAcs[lg];
      v4bf o;
#pragma unroll
      for (int j = 0; j < 4; ++j) {
        const int sg = sg0 + j;
        o[j] = (sg <= lg) ? (bf16)(ga[i][j] * __expf(al - sAcs[sg])) : (bf16)0.f;
      }
      *(v4bf*)&As[r][c4] = o;
    }
#pragma unroll
    for (int i = 0; i < 2; ++i) {
      const int t = threadIdx.x + i*NTHREADS;
      const int r = t >> 4, p4 = (t & 15) * 4;
      Bst[p4+0][r] = (bf16)gb[i][0];
      Bst[p4+1][r] = (bf16)gb[i][1];
      Bst[p4+2][r] = (bf16)gb[i][2];
      Bst[p4+3][r] = (bf16)gb[i][3];
    }
    __syncthreads();
    mma_step(&As[0][0], &Bst[0][0], acc);
    __syncthreads();
  }
  float* C = Y + rowbase*D_INNER + h*HEADDIM;
  store_acc(C, D_INNER, CHUNK, HEADDIM, mb, 0, acc, 0);
}

// ---------------- chunk states S[n,p] = sum_l B[l,n]*decay[l]*X[l,p] ----------------
__global__ void k_states(const float* __restrict__ xBC, const float* __restrict__ Xs,
                         const float* __restrict__ Acs, float* __restrict__ S) {
  __shared__ alignas(16) bf16 As[BM][LDSA];
  __shared__ alignas(16) bf16 Bst[BN][LDSBT];
  __shared__ float sAcs[CHUNK];
  const int zidx = blockIdx.x;
  const int bc = zidx / NHEADS;
  const int h  = zidx % NHEADS;
  for (int t = threadIdx.x; t < CHUNK; t += NTHREADS)
    sAcs[t] = Acs[(long long)zidx*CHUNK + t];
  __syncthreads();
  const float Aend = sAcs[CHUNK - 1];
  const long long rowbase = (long long)bc * CHUNK;
  v8f acc[2][2] = {};
  for (int k0 = 0; k0 < CHUNK; k0 += BK) {
    v4f ga[4], gb[2];
#pragma unroll
    for (int i = 0; i < 4; ++i) {        // A: Bm along n (contiguous), r = l
      const int t = threadIdx.x + i*NTHREADS;
      const int r = t >> 5, n4 = (t & 31) * 4;
      ga[i] = *(const v4f*)&xBC[(rowbase + k0 + r)*CONV_DIM + D_INNER + n4];
    }
#pragma unroll
    for (int i = 0; i < 2; ++i) {
      const int t = threadIdx.x + i*NTHREADS;
      const int r = t >> 4, p4 = (t & 15) * 4;
      gb[i] = *(const v4f*)&Xs[(rowbase + k0 + r)*D_INNER + h*HEADDIM + p4];
    }
#pragma unroll
    for (int i = 0; i < 4; ++i) {
      const int t = threadIdx.x + i*NTHREADS;
      const int r = t >> 5, n4 = (t & 31) * 4;
      const float w = __expf(Aend - sAcs[k0 + r]);
      As[n4+0][r] = (bf16)(ga[i][0] * w);
      As[n4+1][r] = (bf16)(ga[i][1] * w);
      As[n4+2][r] = (bf16)(ga[i][2] * w);
      As[n4+3][r] = (bf16)(ga[i][3] * w);
    }
#pragma unroll
    for (int i = 0; i < 2; ++i) {
      const int t = threadIdx.x + i*NTHREADS;
      const int r = t >> 4, p4 = (t & 15) * 4;
      Bst[p4+0][r] = (bf16)gb[i][0];
      Bst[p4+1][r] = (bf16)gb[i][1];
      Bst[p4+2][r] = (bf16)gb[i][2];
      Bst[p4+3][r] = (bf16)gb[i][3];
    }
    __syncthreads();
    mma_step(&As[0][0], &Bst[0][0], acc);
    __syncthreads();
  }
  float* C = S + (long long)zidx * (D_STATE * HEADDIM);
  store_acc(C, HEADDIM, D_STATE, HEADDIM, 0, 0, acc, 0);
}

// ---------------- inter-chunk recurrence ----------------
__global__ void k_recur(const float* __restrict__ S, const float* __restrict__ Acs,
                        float* __restrict__ Sin) {
  const int b = blockIdx.x / NHEADS;
  const int h = blockIdx.x % NHEADS;
  const int tid = threadIdx.x;
  float r[32];
#pragma unroll
  for (int i = 0; i < 32; ++i) r[i] = 0.f;
  for (int c = 0; c < NCHUNK; ++c) {
    const long long zidx = (long long)(b*NCHUNK + c)*NHEADS + h;
    const long long base = zidx * (D_STATE * HEADDIM);
#pragma unroll
    for (int i = 0; i < 8; ++i)
      *(v4f*)&Sin[base + (tid + i*NTHREADS)*4] = *(const v4f*)&r[i*4];
    const float alpha = __expf(Acs[zidx*CHUNK + CHUNK - 1]);
#pragma unroll
    for (int i = 0; i < 8; ++i) {
      const v4f s = *(const v4f*)&S[base + (tid + i*NTHREADS)*4];
#pragma unroll
      for (int j = 0; j < 4; ++j) r[i*4 + j] = r[i*4 + j]*alpha + s[j];
    }
  }
}

// ---------------- Y += (C .* exp(Acs[l])) @ Sin ----------------
__global__ void k_yoff(const float* __restrict__ xBC, const float* __restrict__ Sin,
                       const float* __restrict__ Acs, float* __restrict__ Y) {
  __shared__ alignas(16) bf16 As[BM][LDSA];
  __shared__ alignas(16) bf16 Bst[BN][LDSBT];
  __shared__ float sAcs[CHUNK];
  const int mtile = blockIdx.x;
  const int zidx  = blockIdx.y;
  const int bc = zidx / NHEADS;
  const int h  = zidx % NHEADS;
  const int mb = mtile * BM;
  for (int t = threadIdx.x; t < CHUNK; t += NTHREADS)
    sAcs[t] = Acs[(long long)zidx*CHUNK + t];
  __syncthreads();
  const long long rowbase = (long long)bc * CHUNK;
  const float* Sb = Sin + (long long)zidx * (D_STATE * HEADDIM);
  v8f acc[2][2] = {};
  for (int k0 = 0; k0 < D_STATE; k0 += BK) {
    v4f ga[4], gb[2];
#pragma unroll
    for (int i = 0; i < 4; ++i) {
      const int t = threadIdx.x + i*NTHREADS;
      const int r = t >> 3, c4 = (t & 7) * 4;
      ga[i] = *(const v4f*)&xBC[(rowbase + mb + r)*CONV_DIM + D_INNER + D_STATE
                                + k0 + c4];
    }
#pragma unroll
    for (int i = 0; i < 2; ++i) {
      const int t = threadIdx.x + i*NTHREADS;
      const int r = t >> 4, p4 = (t & 15) * 4;
      gb[i] = *(const v4f*)&Sb[(k0 + r)*HEADDIM + p4];
    }
#pragma unroll
    for (int i = 0; i < 4; ++i) {
      const int t = threadIdx.x + i*NTHREADS;
      const int r = t >> 3, c4 = (t & 7) * 4;
      const float w = __expf(sAcs[mb + r]);
      v4bf o;
      o[0] = (bf16)(ga[i][0]*w); o[1] = (bf16)(ga[i][1]*w);
      o[2] = (bf16)(ga[i][2]*w); o[3] = (bf16)(ga[i][3]*w);
      *(v4bf*)&As[r][c4] = o;
    }
#pragma unroll
    for (int i = 0; i < 2; ++i) {
      const int t = threadIdx.x + i*NTHREADS;
      const int r = t >> 4, p4 = (t & 15) * 4;
      Bst[p4+0][r] = (bf16)gb[i][0];
      Bst[p4+1][r] = (bf16)gb[i][1];
      Bst[p4+2][r] = (bf16)gb[i][2];
      Bst[p4+3][r] = (bf16)gb[i][3];
    }
    __syncthreads();
    mma_step(&As[0][0], &Bst[0][0], acc);
    __syncthreads();
  }
  float* C = Y + rowbase*D_INNER + h*HEADDIM;
  store_acc(C, D_INNER, CHUNK, HEADDIM, mb, 0, acc, 1);   // accumulate
}

// ---------------- layernorm + SiLU(z) gate ----------------
__global__ void k_norm(float* __restrict__ Y, const float* __restrict__ zxbcdt,
                       const float* __restrict__ norm_w) {
  const int row = blockIdx.x;
  __shared__ float s1[NTHREADS], s2[NTHREADS];
  v4f v[2];
  float a = 0.f, sq = 0.f;
#pragma unroll
  for (int i = 0; i < 2; ++i) {
    v[i] = *(const v4f*)&Y[(long long)row*D_INNER + threadIdx.x*8 + i*4];
#pragma unroll
    for (int j = 0; j < 4; ++j) { a += v[i][j]; sq += v[i][j]*v[i][j]; }
  }
  s1[threadIdx.x] = a; s2[threadIdx.x] = sq;
  __syncthreads();
  for (int off = NTHREADS/2; off > 0; off >>= 1) {
    if (threadIdx.x < off) {
      s1[threadIdx.x] += s1[threadIdx.x + off];
      s2[threadIdx.x] += s2[threadIdx.x + off];
    }
    __syncthreads();
  }
  const float mu  = s1[0] / (float)D_INNER;
  const float var = s2[0] / (float)D_INNER - mu*mu;
  const float inv = rsqrtf(var + 1e-5f);
#pragma unroll
  for (int i = 0; i < 2; ++i) {
    const int idx = threadIdx.x*8 + i*4;
    const v4f z = *(const v4f*)&zxbcdt[(long long)row*D_IN_PROJ + idx];
    const v4f w = *(const v4f*)&norm_w[idx];
    v4f o;
#pragma unroll
    for (int j = 0; j < 4; ++j) {
      const float sz = z[j] / (1.f + expf(-z[j]));
      o[j] = (v[i][j] - mu)*inv*w[j]*sz;
    }
    *(v4f*)&Y[(long long)row*D_INNER + idx] = o;
  }
}

// ---------------- host launcher ----------------
extern "C" void kernel_launch(void* const* d_in, const int* in_sizes, int n_in,
                              void* d_out, int out_size, void* d_ws, size_t ws_size,
                              hipStream_t stream) {
  (void)in_sizes; (void)n_in; (void)out_size; (void)ws_size;
  const float* u       = (const float*)d_in[0];
  const float* W_in    = (const float*)d_in[1];
  const float* conv_w  = (const float*)d_in[2];
  const float* conv_b  = (const float*)d_in[3];
  const float* dt_bias = (const float*)d_in[4];
  const float* A_log   = (const float*)d_in[5];
  const float* norm_w  = (const float*)d_in[6];
  const float* W_out   = (const float*)d_in[7];
  float* out = (float*)d_out;

  float* ws = (float*)d_ws;
  long long off = 0;
  float* zxbcdt = ws + off; off += (long long)ROWS * D_IN_PROJ;
  float* xBC    = ws + off; off += (long long)ROWS * CONV_DIM;
  float* Xs     = ws + off; off += (long long)ROWS * D_INNER;
  float* dA     = ws + off; off += (long long)ROWS * NHEADS;
  float* Acs    = ws + off; off += (long long)BATCH*NCHUNK*NHEADS*CHUNK;
  float* G      = ws + off; off += (long long)BATCH*NCHUNK*CHUNK*CHUNK;
  float* Y      = ws + off; off += (long long)ROWS * D_INNER;
  float* S      = ws + off; off += (long long)BATCH*NCHUNK*NHEADS*D_STATE*HEADDIM;
  float* Sin    = ws + off; off += (long long)BATCH*NCHUNK*NHEADS*D_STATE*HEADDIM;

  const dim3 blk(NTHREADS);

  // 1. in_proj: zxbcdt = u @ W_in   (8192 x 4384, K=1024)
  k_gemm<<<dim3((D_IN_PROJ + BN - 1)/BN, ROWS/BM, 1), blk, 0, stream>>>(
      u, W_in, zxbcdt, ROWS, D_IN_PROJ, D_MODEL, D_MODEL, D_IN_PROJ, D_IN_PROJ,
      0, 0, 0, 0, 0);

  // 2. conv + SiLU + softplus(dt) + dA + dt-scaled X
  k_convact<<<dim3(ROWS), blk, 0, stream>>>(zxbcdt, conv_w, conv_b, dt_bias,
                                            A_log, xBC, Xs, dA);

  // 3. per-(b,c,h) cumsum of dA
  k_cumsum<<<dim3(BATCH*NCHUNK), dim3(32), 0, stream>>>(dA, Acs);

  // 4. G = Cm @ Bm^T per (b,c)  (256 x 256, K=128)
  k_gemm<<<dim3(CHUNK/BN, CHUNK/BM, BATCH*NCHUNK), blk, 0, stream>>>(
      xBC + (D_INNER + D_STATE),   // Cm slice
      xBC + D_INNER,               // Bm slice (transB)
      G, CHUNK, CHUNK, D_STATE, CONV_DIM, CONV_DIM, CHUNK,
      (long long)CHUNK*CONV_DIM, (long long)CHUNK*CONV_DIM,
      (long long)CHUNK*CHUNK, 1, 0);

  // 5. Y_diag per (b,c,h)
  k_ydiag<<<dim3(CHUNK/BM, BATCH*NCHUNK*NHEADS), blk, 0, stream>>>(G, Xs, Acs, Y);

  // 6. chunk states per (b,c,h)
  k_states<<<dim3(BATCH*NCHUNK*NHEADS), blk, 0, stream>>>(xBC, Xs, Acs, S);

  // 7. inter-chunk recurrence
  k_recur<<<dim3(BATCH*NHEADS), blk, 0, stream>>>(S, Acs, Sin);

  // 8. Y += Y_off per (b,c,h)
  k_yoff<<<dim3(CHUNK/BM, BATCH*NCHUNK*NHEADS), blk, 0, stream>>>(xBC, Sin, Acs, Y);

  // 9. layernorm + gate
  k_norm<<<dim3(ROWS), blk, 0, stream>>>(Y, zxbcdt, norm_w);

  // 10. out_proj: out = Y @ W_out  (8192 x 1024, K=2048)
  k_gemm<<<dim3(D_MODEL/BN, ROWS/BM, 1), blk, 0, stream>>>(
      Y, W_out, out, ROWS, D_MODEL, D_INNER, D_INNER, D_MODEL, D_MODEL,
      0, 0, 0, 0, 0);
}
